// DecoderMixer_8718783611324
// MI455X (gfx1250) — matrix operand order repaired and checked by reading, not compiled, not run on hardware
//
#include <hip/hip_runtime.h>
#include <hip/hip_bf16.h>
#include <math.h>

// ---------------- types ----------------
typedef __attribute__((ext_vector_type(16))) __bf16        v16bf;
typedef __attribute__((ext_vector_type(8)))  float         v8f;
typedef __attribute__((ext_vector_type(4)))  float         v4f;
typedef __attribute__((ext_vector_type(4)))  unsigned int  v4u;

struct V4U2 { v4u a, b; };   // 32 bytes -> bit_cast to v16bf

#define N_TOK 4096
#define CDIM  1024
#define LSEQ  13
#define SLAB  ((size_t)N_TOK * CDIM)      // elements in one l-slab
#define WELEM ((size_t)CDIM * CDIM)       // one weight matrix

// ---------------- helpers ----------------
__device__ __forceinline__ unsigned short f2bf(float f) {
  unsigned int u = __builtin_bit_cast(unsigned int, f);
  u += 0x7fffu + ((u >> 16) & 1u);            // round-to-nearest-even
  return (unsigned short)(u >> 16);
}
__device__ __forceinline__ float bf2f(unsigned short h) {
  unsigned int u = ((unsigned int)h) << 16;
  return __builtin_bit_cast(float, u);
}
__device__ __forceinline__ unsigned int pack2(float a, float b) {
  return (unsigned int)f2bf(a) | ((unsigned int)f2bf(b) << 16);
}
// wave-local LDS producer->consumer sync (CDNA5 split DS counter)
__device__ __forceinline__ void wave_lds_sync() {
  __builtin_amdgcn_wave_barrier();
  asm volatile("s_wait_dscnt 0" ::: "memory");
  __builtin_amdgcn_wave_barrier();
}

// =====================================================================
// Kernel 0: one-time f32 -> bf16 conversion (8 values / thread, b128 I/O)
// =====================================================================
__global__ __launch_bounds__(256)
void convert_f32_bf16(const float* __restrict__ src, unsigned short* __restrict__ dst) {
  const size_t base = ((size_t)blockIdx.x * 256 + threadIdx.x) * 8;
  v4f f0 = *(const v4f*)(src + base);
  v4f f1 = *(const v4f*)(src + base + 4);
  v4u p = { pack2(f0.x, f0.y), pack2(f0.z, f0.w),
            pack2(f1.x, f1.y), pack2(f1.z, f1.w) };
  *(v4u*)(dst + base) = p;
}

// =====================================================================
// bf16 WMMA GEMM: out = A(4096x1024) @ W^T(1024x1024), f32 accumulate.
// kind 0: K-proj (+RoPE)  kind 1: V-proj  kind 2: Q-proj (+RoPE, *0.125)
// kind 3: O-proj (f32 out to d_out)
// WG tile 256(M) x 64(N), K-step 64; 8 waves, wave tile 32(M) x 64(N).
// Register double-buffering: tile i+1 is loaded to VGPRs while tile i
// is consumed by 16 wmma, hiding global-load latency without extra LDS.
// =====================================================================
__global__ __launch_bounds__(256)
void gemm_kernel(const unsigned short* __restrict__ Xb,
                 const unsigned short* __restrict__ Wqb,
                 const unsigned short* __restrict__ Wkb,
                 const unsigned short* __restrict__ Wvb,
                 const unsigned short* __restrict__ Wob,
                 const unsigned short* __restrict__ Ain,   // O-proj activations
                 unsigned short* __restrict__ Kb,
                 unsigned short* __restrict__ Vb,
                 unsigned short* __restrict__ Qb,
                 float* __restrict__ outF,
                 int kind_override)
{
  __shared__ __align__(16) unsigned short smA[256 * 72];  // 256 rows x 64 k (pad->72)
  __shared__ __align__(16) unsigned short smB[64 * 72];   //  64 rows x 64 k (pad->72)

  const int tid  = threadIdx.x;
  const int w    = tid >> 5;          // wave 0..7
  const int lane = tid & 31;
  const int ln   = lane & 15;
  const int hi   = lane >> 4;

  const int z = blockIdx.z;
  int kind, l = 0;
  const unsigned short* Ag;
  const unsigned short* Bg;
  unsigned short*       outB = nullptr;

  if (kind_override == 3)  { kind = 3;             Ag = Ain;                    Bg = Wob; }
  else if (z < 13)         { kind = 0; l = z;      Ag = Xb + (size_t)z * SLAB;  Bg = Wkb; outB = Kb + (size_t)z * SLAB; }
  else if (z < 26)         { kind = 1; l = z - 13; Ag = Xb + (size_t)l * SLAB;  Bg = Wvb; outB = Vb + (size_t)l * SLAB; }
  else                     { kind = 2; l = 12;     Ag = Xb + (size_t)12 * SLAB; Bg = Wqb; outB = Qb; }

  const int m_base = blockIdx.y * 256;
  const int n_base = blockIdx.x * 64;

  // staging addresses: row_i = r0 + 32*i, column chunk c8 constant per thread
  const int r0 = tid >> 3;
  const int c8 = (tid & 7) * 8;
  const unsigned short* agp = Ag + (size_t)(m_base + r0) * CDIM + c8;
  const unsigned short* bgp = Bg + (size_t)(n_base + r0) * CDIM + c8;
  unsigned short* alp = smA + r0 * 72 + c8;
  unsigned short* blp = smB + r0 * 72 + c8;

  v8f acc[2][4];
  #pragma unroll
  for (int mi = 0; mi < 2; mi++)
    #pragma unroll
    for (int t = 0; t < 4; t++)
      acc[mi][t] = (v8f){0.f,0.f,0.f,0.f,0.f,0.f,0.f,0.f};

  // preload tile k0 = 0 into registers
  v4u abuf[8], bbuf[2];
  #pragma unroll
  for (int i = 0; i < 8; i++) abuf[i] = *(const v4u*)(agp + (size_t)i * 32 * CDIM);
  #pragma unroll
  for (int i = 0; i < 2; i++) bbuf[i] = *(const v4u*)(bgp + (size_t)i * 32 * CDIM);

  for (int k0 = 0; k0 < CDIM; k0 += 64) {
    // ---- commit staged registers to LDS ----
    #pragma unroll
    for (int i = 0; i < 8; i++) *(v4u*)(alp + i * 32 * 72) = abuf[i];
    #pragma unroll
    for (int i = 0; i < 2; i++) *(v4u*)(blp + i * 32 * 72) = bbuf[i];
    __syncthreads();

    // ---- issue next tile's global loads (wraps on last iter; harmless) ----
    const int kn = (k0 + 64) & (CDIM - 1);
    #pragma unroll
    for (int i = 0; i < 8; i++) abuf[i] = *(const v4u*)(agp + (size_t)i * 32 * CDIM + kn);
    #pragma unroll
    for (int i = 0; i < 2; i++) bbuf[i] = *(const v4u*)(bgp + (size_t)i * 32 * CDIM + kn);
    // prefetch two tiles ahead into WGP caches
    __builtin_prefetch(agp + ((k0 + 128) & (CDIM - 1)), 0, 3);

    // ---- 16 wmma from LDS ----
    #pragma unroll
    for (int j = 0; j < 2; j++) {          // two 32-wide k sub-steps
      v16bf af[2];
      #pragma unroll
      for (int mi = 0; mi < 2; mi++) {
        const int arow = w * 32 + mi * 16 + ln;
        V4U2 ap;
        ap.a = *(const v4u*)(smA + arow * 72 + j * 32 + hi * 8);
        ap.b = *(const v4u*)(smA + arow * 72 + j * 32 + 16 + hi * 8);
        af[mi] = __builtin_bit_cast(v16bf, ap);
      }
      #pragma unroll
      for (int t = 0; t < 4; t++) {
        const int brow = t * 16 + ln;
        V4U2 bp;
        bp.a = *(const v4u*)(smB + brow * 72 + j * 32 + hi * 8);
        bp.b = *(const v4u*)(smB + brow * 72 + j * 32 + 16 + hi * 8);
        v16bf bfrag = __builtin_bit_cast(v16bf, bp);
        #pragma unroll
        for (int mi = 0; mi < 2; mi++) {
          acc[mi][t] = __builtin_amdgcn_wmma_f32_16x16x32_bf16(
                         false, af[mi], false, bfrag, (short)0, acc[mi][t], false, false);
        }
      }
    }
    __syncthreads();
  }

  // ---- epilogue: RoPE (K/Q), scale (Q), store ----
  // D layout: vgpr r, lanes 0-15 -> m=r, n=lane; lanes 16-31 -> m=8+r, n=lane-16
  const float rope_l = (float)l;
  #pragma unroll
  for (int t = 0; t < 4; t++) {
    const int d = t * 16 + ln;          // column within 64-wide head
    float cv = 1.f, sv = 0.f, sg = 1.f;
    if (kind == 0 || kind == 2) {
      float ang = rope_l * __powf(10000.f, -(float)(d & 31) * (1.f / 32.f));
      cv = __cosf(ang); sv = __sinf(ang);
      sg = (d < 32) ? -1.f : 1.f;       // rotate-half sign
    }
    #pragma unroll
    for (int mi = 0; mi < 2; mi++) {
      #pragma unroll
      for (int r = 0; r < 8; r++) {
        const int    grow = m_base + w * 32 + mi * 16 + hi * 8 + r;
        const size_t off  = (size_t)grow * CDIM + n_base + d;
        float x = acc[mi][t][r];
        if (kind == 3) {
          outF[off] = x;
        } else {
          float val = x;
          if (kind != 1) {                       // RoPE partner d^32 -> acc[mi][t^2]
            float y = acc[mi][t ^ 2][r];
            val = x * cv + sg * y * sv;
            if (kind == 2) val *= 0.125f;        // 1/sqrt(64) folded into q
          }
          outB[off] = f2bf(val);
        }
      }
    }
  }
}

// =====================================================================
// Attention: one wave per token. Lanes 0..12 own the 13 K/V rows per head.
// =====================================================================
__global__ __launch_bounds__(256)
void attn_kernel(const unsigned short* __restrict__ Kb,
                 const unsigned short* __restrict__ Vb,
                 const unsigned short* __restrict__ Qb,
                 unsigned short* __restrict__ Ab)
{
  __shared__ __align__(16) float qsh[8][CDIM];       // 32 KB
  __shared__ __align__(16) float vsh[8][LSEQ][68];   // padded rows
  __shared__ float ssh[8][16];

  const int tid  = threadIdx.x;
  const int w    = tid >> 5;
  const int lane = tid & 31;
  const int n    = blockIdx.x * 8 + w;               // grid = 512 -> n in [0,4096)

  const unsigned short* qr = Qb + (size_t)n * CDIM;
  for (int i = lane; i < CDIM; i += 32) qsh[w][i] = bf2f(qr[i]);
  wave_lds_sync();

  for (int h = 0; h < 16; h++) {
    if (lane < LSEQ) {
      const unsigned short* kr = Kb + (size_t)lane * SLAB + (size_t)n * CDIM + h * 64;
      const unsigned short* vr = Vb + (size_t)lane * SLAB + (size_t)n * CDIM + h * 64;
      float s = 0.f;
      #pragma unroll
      for (int i = 0; i < 64; i++) {
        s += bf2f(kr[i]) * qsh[w][h * 64 + i];
        vsh[w][lane][i] = bf2f(vr[i]);
      }
      ssh[w][lane] = s;
    }
    wave_lds_sync();

    float p[LSEQ], mx = -3.0e38f, den = 0.f;
    #pragma unroll
    for (int j = 0; j < LSEQ; j++) { p[j] = ssh[w][j]; mx = fmaxf(mx, p[j]); }
    #pragma unroll
    for (int j = 0; j < LSEQ; j++) { p[j] = __expf(p[j] - mx); den += p[j]; }
    const float inv = 1.f / den;

    float o0 = 0.f, o1 = 0.f;                        // lane owns dims 2*lane, 2*lane+1
    #pragma unroll
    for (int j = 0; j < LSEQ; j++) {
      o0 += p[j] * vsh[w][j][2 * lane];
      o1 += p[j] * vsh[w][j][2 * lane + 1];
    }
    o0 *= inv; o1 *= inv;
    *(unsigned int*)(Ab + (size_t)n * CDIM + h * 64 + 2 * lane) = pack2(o0, o1);
    wave_lds_sync();
  }
}

// =====================================================================
extern "C" void kernel_launch(void* const* d_in, const int* in_sizes, int n_in,
                              void* d_out, int out_size, void* d_ws, size_t ws_size,
                              hipStream_t stream) {
  (void)in_sizes; (void)n_in; (void)out_size; (void)ws_size;
  const float* X  = (const float*)d_in[0];   // (13, 4, 1024, 1024)
  const float* Wq = (const float*)d_in[1];
  const float* Wk = (const float*)d_in[2];
  const float* Wv = (const float*)d_in[3];
  const float* Wo = (const float*)d_in[4];
  float* out = (float*)d_out;                // (4, 1024, 1024) f32

  unsigned short* Xb  = (unsigned short*)d_ws;         // bf16 copy of layer_outputs
  unsigned short* Wqb = Xb  + (size_t)LSEQ * SLAB;
  unsigned short* Wkb = Wqb + WELEM;
  unsigned short* Wvb = Wkb + WELEM;
  unsigned short* Wob = Wvb + WELEM;
  unsigned short* Kb  = Wob + WELEM;                   // roped K   [13][4096][1024]
  unsigned short* Vb  = Kb  + (size_t)LSEQ * SLAB;     // V         [13][4096][1024]
  unsigned short* Qb  = Vb  + (size_t)LSEQ * SLAB;     // roped q(l=12) [4096][1024]
  unsigned short* Ab  = Qb  + SLAB;                    // attn out  [4096][1024]

  dim3 blk(256);
  // one-time bf16 conversions (each elem converted exactly once)
  convert_f32_bf16<<<dim3((unsigned)((LSEQ * SLAB) / 2048)), blk, 0, stream>>>(X, Xb);
  convert_f32_bf16<<<dim3((unsigned)(WELEM / 2048)), blk, 0, stream>>>(Wq, Wqb);
  convert_f32_bf16<<<dim3((unsigned)(WELEM / 2048)), blk, 0, stream>>>(Wk, Wkb);
  convert_f32_bf16<<<dim3((unsigned)(WELEM / 2048)), blk, 0, stream>>>(Wv, Wvb);
  convert_f32_bf16<<<dim3((unsigned)(WELEM / 2048)), blk, 0, stream>>>(Wo, Wob);

  // 27 slabs: 13x K-proj, 13x V-proj, 1x Q-proj (l=12). RoPE fused in epilogue.
  gemm_kernel<<<dim3(16, 16, 27), blk, 0, stream>>>(
      Xb, Wqb, Wkb, Wvb, Wob, nullptr, Kb, Vb, Qb, nullptr, -1);
  attn_kernel<<<dim3(512), blk, 0, stream>>>(Kb, Vb, Qb, Ab);
  // output projection: (4096x1024) bf16 @ Wo^T -> f32 d_out
  gemm_kernel<<<dim3(16, 16, 1), blk, 0, stream>>>(
      Xb, Wqb, Wkb, Wvb, Wob, Ab, Kb, Vb, Qb, out, 3);
}
